// GNN_39745627357498
// MI455X (gfx1250) — matrix operand order repaired and checked by reading, not compiled, not run on hardware
//
#include <hip/hip_runtime.h>
#include <hip/hip_bf16.h>
#include <stdint.h>
#include <stddef.h>

// ---------------------------------------------------------------------------
// Problem constants (from reference)
// ---------------------------------------------------------------------------
#define NNODES 4096
#define NEDGES 65536
#define EP     (NEDGES + NNODES)   // 69632 edges incl. self loops
#define FEAT   16384
#define SLOPE  0.2f
#define NUMCLS 20

typedef __attribute__((ext_vector_type(16))) __bf16 v16bf;
typedef __attribute__((ext_vector_type(8)))  float  v8f;
typedef __attribute__((ext_vector_type(4)))  unsigned int v4u;
typedef __attribute__((ext_vector_type(4)))  int v4i;
typedef __attribute__((ext_vector_type(8)))  int v8i;

union FragBF {
    uint4          q[2];
    unsigned short us[16];
    v16bf          v;
};

// ---------------------------------------------------------------------------
// Device helpers
// ---------------------------------------------------------------------------
__device__ __forceinline__ unsigned short f32_to_bf16_rne(float f) {
    unsigned int u = __float_as_uint(f);
    u += 0x7FFFu + ((u >> 16) & 1u);   // round-to-nearest-even
    return (unsigned short)(u >> 16);
}

__device__ __forceinline__ void edge_sd(int e, const int* __restrict__ ei, int& s, int& d) {
    if (e < NEDGES) { s = ei[e]; d = ei[NEDGES + e]; }
    else            { s = e - NEDGES; d = s; }       // self loop
}

// float atomic max via signed/unsigned int trick (compile-safe everywhere)
__device__ __forceinline__ void atomic_max_f32(float* addr, float val) {
    if (val >= 0.0f) atomicMax((int*)addr, __float_as_int(val));
    else             atomicMin((unsigned int*)addr, __float_as_uint(val));
}

// ---------------------------------------------------------------------------
// Utility kernels
// ---------------------------------------------------------------------------
__global__ void k_fill_f32(float* p, long n, float v) {
    long i  = (long)blockIdx.x * blockDim.x + threadIdx.x;
    long st = (long)gridDim.x * blockDim.x;
    for (; i < n; i += st) p[i] = v;
}

__global__ void k_f32_to_bf16(const float* __restrict__ in, unsigned short* __restrict__ out, long n) {
    long i  = (long)blockIdx.x * blockDim.x + threadIdx.x;
    long st = (long)gridDim.x * blockDim.x;
    for (; i < n; i += st) out[i] = f32_to_bf16_rne(in[i]);
}

// ---------------------------------------------------------------------------
// CSR build (in-edges per destination node)
// ---------------------------------------------------------------------------
__global__ void k_deg(const int* __restrict__ ei, int* __restrict__ deg) {
    int e = blockIdx.x * blockDim.x + threadIdx.x;
    if (e >= EP) return;
    int s, d; edge_sd(e, ei, s, d);
    atomicAdd(&deg[d], 1);
}

__global__ void k_scan(const int* __restrict__ deg, int* __restrict__ rowst) {
    if (blockIdx.x == 0 && threadIdx.x == 0) {
        int acc = 0;
        for (int i = 0; i < NNODES; ++i) { rowst[i] = acc; acc += deg[i]; }
        rowst[NNODES] = acc;
    }
}

__global__ void k_fill_csr(const int* __restrict__ ei, const int* __restrict__ rowst,
                           int* __restrict__ cursor, int* __restrict__ csr_src,
                           int* __restrict__ csr_eid) {
    int e = blockIdx.x * blockDim.x + threadIdx.x;
    if (e >= EP) return;
    int s, d; edge_sd(e, ei, s, d);
    int p    = atomicAdd(&cursor[d], 1);
    int slot = rowst[d] + p;
    csr_src[slot] = s;
    csr_eid[slot] = e;
}

// ---------------------------------------------------------------------------
// Attention pipeline (generic over H heads, C channels/head; head-major cols)
// ---------------------------------------------------------------------------
__global__ void k_logits(const float* __restrict__ xl, const float* __restrict__ xr,
                         const float* __restrict__ att, const int* __restrict__ ei,
                         float* __restrict__ logits, float* __restrict__ mmax,
                         int H, int C) {
    int wid  = blockIdx.x * (blockDim.x >> 5) + (threadIdx.x >> 5);
    int lane = threadIdx.x & 31;
    if (wid >= EP * H) return;
    int e = wid / H;
    int h = wid - e * H;
    int s, d; edge_sd(e, ei, s, d);
    long HC = (long)H * C;
    const float* pl = xl + (long)s * HC + (long)h * C;
    const float* pr = xr + (long)d * HC + (long)h * C;
    const float* pa = att + (long)h * C;
    float part = 0.0f;
    for (int c = lane; c < C; c += 32) {
        float v = pl[c] + pr[c];
        v = (v > 0.0f) ? v : SLOPE * v;        // leaky_relu(0.2)
        part += v * pa[c];
    }
    for (int off = 16; off > 0; off >>= 1) part += __shfl_xor(part, off, 32);
    if (lane == 0) {
        logits[(long)e * H + h] = part;
        atomic_max_f32(&mmax[(long)d * H + h], part);
    }
}

__global__ void k_expw(float* __restrict__ logits, const float* __restrict__ mmax,
                       float* __restrict__ denom, const int* __restrict__ ei, int H) {
    long i = (long)blockIdx.x * blockDim.x + threadIdx.x;
    if (i >= (long)EP * H) return;
    int e = (int)(i / H);
    int h = (int)(i - (long)e * H);
    int s, d; edge_sd(e, ei, s, d);
    float w = __expf(logits[i] - mmax[(long)d * H + h]);
    logits[i] = w;
    atomicAdd(&denom[(long)d * H + h], w);
}

__global__ void k_aggregate(const float* __restrict__ xl, const float* __restrict__ w,
                            const float* __restrict__ denom, const float* __restrict__ bias,
                            const int* __restrict__ rowst, const int* __restrict__ csr_src,
                            const int* __restrict__ csr_eid,
                            float* __restrict__ out, int H, int C) {
    int  i   = blockIdx.y;
    long HC  = (long)H * C;
    int  col = blockIdx.x * blockDim.x + threadIdx.x;
    if (col >= HC) return;
    int   h     = col / C;
    float inv_d = 1.0f / denom[(long)i * H + h];
    float acc   = bias[col];
    int p0 = rowst[i], p1 = rowst[i + 1];
    for (int p = p0; p < p1; ++p) {
        int s = csr_src[p];
        int e = csr_eid[p];
        float alpha = w[(long)e * H + h] * inv_d;
        acc += alpha * xl[(long)s * HC + col];
    }
    out[(long)i * HC + col] = acc;
}

// ---------------------------------------------------------------------------
// Small dense matmuls (K too small for WMMA)
// ---------------------------------------------------------------------------
__global__ void k_mm_naive(const float* __restrict__ A, const float* __restrict__ B,
                           float* __restrict__ Cm, int M, int K, int Nn) {
    long i = (long)blockIdx.x * blockDim.x + threadIdx.x;
    if (i >= (long)M * Nn) return;
    int m = (int)(i / Nn);
    int n = (int)(i - (long)m * Nn);
    const float* a = A + (long)m * K;
    float acc = 0.0f;
    for (int k = 0; k < K; ++k) acc += a[k] * B[(long)k * Nn + n];
    Cm[i] = acc;
}

__global__ void k_mm_heads(const float* __restrict__ x1, const float* __restrict__ Wh,
                           float* __restrict__ out) {
    long i = (long)blockIdx.x * blockDim.x + threadIdx.x;
    if (i >= (long)NNODES * FEAT) return;
    int m    = (int)(i >> 14);
    int col  = (int)(i & (FEAT - 1));
    int conv = col >> 10;
    int c    = col & 1023;
    const float* a  = x1 + (long)m * 16;
    const float* wp = Wh + ((long)conv * 16) * 1024 + c;
    float acc = 0.0f;
#pragma unroll
    for (int k = 0; k < 16; ++k) acc += a[k] * wp[(long)k * 1024];
    out[i] = acc;
}

// ---------------------------------------------------------------------------
// Big GEMM: C[M,N] = A[M,K](bf16) @ B[K,N](bf16), fp32 accumulate.
// Block = 256 threads (8 waves), tile 128(M) x 128(N) x 32(K).
// Wave grid 4x2, each wave: 2x4 of v_wmma_f32_16x16x32_bf16 (8 WMMA / K-step).
// A tile staged into padded LDS by the Tensor Data Mover (tensor_load_to_lds,
// pad_interval=64B pad_amount=16B -> 80B pitch), B tile transposed manually.
// ---------------------------------------------------------------------------
#define BM 128
#define BN 128
#define BK 32
#define LDSPITCH 40   // ushorts per row; 80B pitch, 16B-aligned, conflict-light

__global__ __launch_bounds__(256)
void k_gemm_bf16(const unsigned short* __restrict__ A, const unsigned short* __restrict__ B,
                 float* __restrict__ Cm, int M, int K, int Nn) {
    __shared__ __align__(16) unsigned short As[BM][LDSPITCH];
    __shared__ __align__(16) unsigned short Bs[BN][LDSPITCH];   // transposed: Bs[n][k]

    int tid  = threadIdx.x;
    int m0   = blockIdx.y * BM;
    int n0   = blockIdx.x * BN;
    int wave = tid >> 5;
    int lane = tid & 31;
    int wm   = (wave & 3) * 32;     // wave row base within block tile
    int wn   = (wave >> 2) * 64;    // wave col base within block tile
    int half = lane >> 4;
    int l16  = lane & 15;

    v8f acc[2][4];
#pragma unroll
    for (int r = 0; r < 2; ++r)
#pragma unroll
        for (int c = 0; c < 4; ++c) acc[r][c] = (v8f){0.f,0.f,0.f,0.f,0.f,0.f,0.f,0.f};

    // --- TDM descriptor pieces (wave-uniform) for the A tile: BK x BM rows ---
    const unsigned int       lds_base = (unsigned int)(size_t)&As[0][0];
    const unsigned long long gaBase   = (unsigned long long)(size_t)A + (size_t)m0 * K * 2ull;
    // group1 dword0: data_size=2B (code 1), pad_enable=1, pad_interval=3 (64B),
    //                pad_amount=3 (4 dwords = 16B)  => LDS pitch 80B
    const unsigned int g1d0 = (1u << 16) | (1u << 20) | (3u << 22) | (3u << 25);

    const int ktiles = K / BK;
    for (int kt = 0; kt < ktiles; ++kt) {
        __syncthreads();   // previous stage fully consumed

        if (wave == 0) {
            unsigned long long ga = gaBase + (unsigned long long)kt * BK * 2ull;
            v4u g0;
            g0.x = 1u;                                   // count=1, user mode
            g0.y = lds_base;                             // LDS byte address
            g0.z = (unsigned int)ga;                     // global addr [31:0]
            g0.w = (unsigned int)((ga >> 32) & 0x01FFFFFFull) | 0x80000000u; // [56:32] | type=2
            v8i g1;
            g1[0] = (int)g1d0;
            g1[1] = (int)((K & 0xFFFF) << 16);                          // tensor_dim0 lo16
            g1[2] = (int)(((K >> 16) & 0xFFFF) | ((M & 0xFFFF) << 16)); // dim0 hi | dim1 lo
            g1[3] = (int)(((M >> 16) & 0xFFFF) | (BK << 16));           // dim1 hi | tile_dim0
            g1[4] = BM;                                                 // tile_dim1 (tile_dim2=0)
            g1[5] = K;                                                  // tensor_dim0_stride lo32
            g1[6] = 0;
            g1[7] = 0;
            v4i gz = {0, 0, 0, 0};
#if defined(__clang_major__) && (__clang_major__ >= 23)
            v8i gz8 = {0, 0, 0, 0, 0, 0, 0, 0};
            __builtin_amdgcn_tensor_load_to_lds(g0, g1, gz, gz, gz8, 0);
#else
            __builtin_amdgcn_tensor_load_to_lds(g0, g1, gz, gz, 0);
#endif
        }

        // --- B tile: 32(K) x 128(N), uint4 global loads, transposed LDS stores ---
        {
            const uint4* Bg    = (const uint4*)(B + (size_t)(kt * BK) * Nn + n0);
            const int    qRow  = Nn >> 3;                 // uint4 per global row
            for (int i = tid; i < BK * (BN / 8); i += 256) {   // 32*16 = 512
                int k  = i >> 4;
                int nq = i & 15;
                union { uint4 q; unsigned short us[8]; } u;
                u.q = Bg[(size_t)k * qRow + nq];
#pragma unroll
                for (int j = 0; j < 8; ++j) Bs[nq * 8 + j][k] = u.us[j];
            }
        }

        if (wave == 0) __builtin_amdgcn_s_wait_tensorcnt(0);
        __syncthreads();   // A (TDM) + B (manual) visible to all waves

        // --- fragment loads matching ISA 16-bit WMMA layouts ---
        FragBF a0, a1, bfr[4];
        {
            const unsigned short* pr = &As[wm + l16][half * 8];
            a0.q[0] = *(const uint4*)pr;  a0.q[1] = *(const uint4*)(pr + 16);
            pr = &As[wm + 16 + l16][half * 8];
            a1.q[0] = *(const uint4*)pr;  a1.q[1] = *(const uint4*)(pr + 16);
#pragma unroll
            for (int tc = 0; tc < 4; ++tc) {
                pr = &Bs[wn + tc * 16 + l16][half * 8];
                bfr[tc].q[0] = *(const uint4*)pr;
                bfr[tc].q[1] = *(const uint4*)(pr + 16);
            }
        }
#pragma unroll
        for (int tc = 0; tc < 4; ++tc) {
            acc[0][tc] = __builtin_amdgcn_wmma_f32_16x16x32_bf16(false, a0.v, false, bfr[tc].v,
                                                                 (short)0, acc[0][tc], false, false);
            acc[1][tc] = __builtin_amdgcn_wmma_f32_16x16x32_bf16(false, a1.v, false, bfr[tc].v,
                                                                 (short)0, acc[1][tc], false, false);
        }
    }

    // --- store: C/D layout row = v + 8*half, col = l16 ---
    int crow = m0 + wm + half * 8;
    int ccol = n0 + wn + l16;
#pragma unroll
    for (int tr = 0; tr < 2; ++tr)
#pragma unroll
        for (int tc = 0; tc < 4; ++tc)
#pragma unroll
            for (int v = 0; v < 8; ++v)
                Cm[(size_t)(crow + tr * 16 + v) * Nn + (ccol + tc * 16)] = acc[tr][tc][v];
}

// ---------------------------------------------------------------------------
// Output layer: out[row, c] = g[row,:] @ W_out[:, c] + b_out[c]
// ---------------------------------------------------------------------------
__global__ void k_out_linear(const float* __restrict__ g, const float* __restrict__ W,
                             const float* __restrict__ b, float* __restrict__ out) {
    __shared__ float red[NUMCLS];
    int row = blockIdx.x;
    int tid = threadIdx.x;
    if (tid < NUMCLS) red[tid] = 0.0f;
    __syncthreads();
    const float* gr = g + (size_t)row * FEAT;
    float acc[NUMCLS];
#pragma unroll
    for (int c = 0; c < NUMCLS; ++c) acc[c] = 0.0f;
    for (int k = tid; k < FEAT; k += blockDim.x) {
        float gv = gr[k];
        const float* wk = W + (size_t)k * NUMCLS;
#pragma unroll
        for (int c = 0; c < NUMCLS; ++c) acc[c] += gv * wk[c];
    }
#pragma unroll
    for (int c = 0; c < NUMCLS; ++c) atomicAdd(&red[c], acc[c]);
    __syncthreads();
    if (tid < NUMCLS) out[(size_t)row * NUMCLS + tid] = red[tid] + b[tid];
}

// ---------------------------------------------------------------------------
// Host-side: generic GATv2 attention layer driver
// ---------------------------------------------------------------------------
static void run_attention(const float* xl, const float* xr, const float* att, const float* bias,
                          float* out, int H, int C, const int* ei,
                          float* wlog, float* mbuf, float* dbuf,
                          const int* rowst, const int* csr_src, const int* csr_eid,
                          hipStream_t stream) {
    long EH = (long)EP * H;
    k_fill_f32<<<256, 256, 0, stream>>>(mbuf, (long)NNODES * H, -1.0e30f);
    hipMemsetAsync(dbuf, 0, (size_t)NNODES * H * sizeof(float), stream);
    int blocks = (int)((EH + 7) / 8);               // 8 waves / block
    k_logits<<<blocks, 256, 0, stream>>>(xl, xr, att, ei, wlog, mbuf, H, C);
    k_expw<<<(int)((EH + 255) / 256), 256, 0, stream>>>(wlog, mbuf, dbuf, ei, H);
    long HC = (long)H * C;
    dim3 gridA((unsigned)((HC + 255) / 256), NNODES);
    k_aggregate<<<gridA, 256, 0, stream>>>(xl, wlog, dbuf, bias, rowst, csr_src, csr_eid, out, H, C);
}

// ---------------------------------------------------------------------------
// Entry point
// ---------------------------------------------------------------------------
extern "C" void kernel_launch(void* const* d_in, const int* in_sizes, int n_in,
                              void* d_out, int out_size, void* d_ws, size_t ws_size,
                              hipStream_t stream) {
    (void)in_sizes; (void)n_in; (void)out_size; (void)ws_size;

    const float* x         = (const float*)d_in[0];
    const int*   ei        = (const int*)  d_in[1];
    const float* W_init_l  = (const float*)d_in[2];
    const float* W_init_r  = (const float*)d_in[3];
    const float* att_init  = (const float*)d_in[4];
    const float* b_init    = (const float*)d_in[5];
    const float* W_heads_l = (const float*)d_in[6];
    const float* W_heads_r = (const float*)d_in[7];
    const float* att_heads = (const float*)d_in[8];
    const float* b_heads   = (const float*)d_in[9];
    const float* W_gat_l   = (const float*)d_in[10];
    const float* W_gat_r   = (const float*)d_in[11];
    const float* att_gat   = (const float*)d_in[12];
    const float* b_gat     = (const float*)d_in[13];
    const float* W_out     = (const float*)d_in[14];
    const float* b_out     = (const float*)d_in[15];

    // bump allocator over workspace
    size_t off = 0;
    auto alloc = [&](size_t bytes) -> void* {
        void* p = (char*)d_ws + off;
        off += (bytes + 255) & ~(size_t)255;
        return p;
    };
    float*          xl_big  = (float*)alloc((size_t)NNODES * FEAT * 4);   // mid xl, then gat xl
    float*          xr_big  = (float*)alloc((size_t)NNODES * FEAT * 4);   // mid xr, then gat xr
    float*          hbuf    = (float*)alloc((size_t)NNODES * FEAT * 4);   // h, then g
    unsigned short* hbf     = (unsigned short*)alloc((size_t)NNODES * FEAT * 2);
    unsigned short* wbf     = (unsigned short*)alloc((size_t)FEAT * FEAT * 2);
    float*          x1      = (float*)alloc((size_t)NNODES * 16 * 4);
    float*          xl_i    = (float*)alloc((size_t)NNODES * 16 * 4);
    float*          xr_i    = (float*)alloc((size_t)NNODES * 16 * 4);
    float*          wlog    = (float*)alloc((size_t)EP * 16 * 4);
    float*          mbuf    = (float*)alloc((size_t)NNODES * 16 * 4);
    float*          dbuf    = (float*)alloc((size_t)NNODES * 16 * 4);
    int*            deg     = (int*)alloc((size_t)NNODES * 4);
    int*            rowst   = (int*)alloc((size_t)(NNODES + 1) * 4);
    int*            cursor  = (int*)alloc((size_t)NNODES * 4);
    int*            csr_src = (int*)alloc((size_t)EP * 4);
    int*            csr_eid = (int*)alloc((size_t)EP * 4);

    // --- CSR of in-edges (shared by all three conv layers) ---
    hipMemsetAsync(deg,    0, (size_t)NNODES * 4, stream);
    hipMemsetAsync(cursor, 0, (size_t)NNODES * 4, stream);
    k_deg<<<(EP + 255) / 256, 256, 0, stream>>>(ei, deg);
    k_scan<<<1, 1, 0, stream>>>(deg, rowst);
    k_fill_csr<<<(EP + 255) / 256, 256, 0, stream>>>(ei, rowst, cursor, csr_src, csr_eid);

    // --- init conv: GATv2(64 -> 8, heads=2) ---
    k_mm_naive<<<(NNODES * 16 + 255) / 256, 256, 0, stream>>>(x, W_init_l, xl_i, NNODES, 64, 16);
    k_mm_naive<<<(NNODES * 16 + 255) / 256, 256, 0, stream>>>(x, W_init_r, xr_i, NNODES, 64, 16);
    run_attention(xl_i, xr_i, att_init, b_init, x1, 2, 8, ei,
                  wlog, mbuf, dbuf, rowst, csr_src, csr_eid, stream);

    // --- 16 x GATv2(16 -> 1024, heads=1) == one 16-head conv, head-major cols ---
    long tot = (long)NNODES * FEAT;
    k_mm_heads<<<(int)((tot + 255) / 256), 256, 0, stream>>>(x1, W_heads_l, xl_big);
    k_mm_heads<<<(int)((tot + 255) / 256), 256, 0, stream>>>(x1, W_heads_r, xr_big);
    run_attention(xl_big, xr_big, att_heads, b_heads, hbuf, 16, 1024, ei,
                  wlog, mbuf, dbuf, rowst, csr_src, csr_eid, stream);

    // --- gat conv: xl/xr = h @ W_gat_{l,r}  (4.4 TFLOP, WMMA bf16 path) ---
    k_f32_to_bf16<<<4096, 256, 0, stream>>>(hbuf, hbf, tot);
    dim3 ggrid(FEAT / BN, NNODES / BM);
    k_f32_to_bf16<<<8192, 256, 0, stream>>>(W_gat_l, wbf, (long)FEAT * FEAT);
    k_gemm_bf16<<<ggrid, 256, 0, stream>>>(hbf, wbf, xl_big, NNODES, FEAT, FEAT);
    k_f32_to_bf16<<<8192, 256, 0, stream>>>(W_gat_r, wbf, (long)FEAT * FEAT);
    k_gemm_bf16<<<ggrid, 256, 0, stream>>>(hbf, wbf, xr_big, NNODES, FEAT, FEAT);
    run_attention(xl_big, xr_big, att_gat, b_gat, hbuf, 16, 1024, ei,
                  wlog, mbuf, dbuf, rowst, csr_src, csr_eid, stream);

    // --- output linear: g @ W_out + b_out ---
    k_out_linear<<<NNODES, 256, 0, stream>>>(hbuf, W_out, b_out, (float*)d_out);
}
